// MCDModel_17609365913634
// MI455X (gfx1250) — matrix operand order repaired and checked by reading, not compiled, not run on hardware
//
#include <hip/hip_runtime.h>
#include <hip/hip_bf16.h>

// ---------------------------------------------------------------------------
// MI455X (gfx1250) implementation.
// Precision strategy: fp32 inputs staged to LDS as bf16, all large GEMMs run
// on v_wmma_f32_16x16x32_bf16 (f32 accumulate). Attention cores / GRU scan /
// gating are VALU (tiny or sequential).
// GEMM: 128x128 block tile, K-step 32, double-buffered LDS, branch-free
// vectorized staging (b128 global loads), one barrier per K-step.
// NOTE: all GEMM call sites use K % 32 == 0 (the kernel assumes it).
// ---------------------------------------------------------------------------

typedef __attribute__((ext_vector_type(16))) __bf16 bf16x16;
typedef __attribute__((ext_vector_type(8)))  float  f32x8;
typedef __attribute__((ext_vector_type(4)))  unsigned int u32x4;

#define LDSK 40  // padded LDS row stride (bf16 elems); 80B rows keep 16B align

union FragU { u32x4 u[2]; bf16x16 v; };
union Pack4 { __bf16 h[4]; ushort4 s; };

#define ACT_NONE 0
#define ACT_RELU 1
#define ACT_SIG  2

// C[M,N] = act(A[M,K] @ op(B) + bias + R)
// TRANSB=1: B is [N,K] row-major (torch Linear).  TRANSB=0: B is [K,N].
// gridDim.z batches (MoE expert banks) via sA/sB/sC/sBias strides.
template<int TRANSB>
__global__ __launch_bounds__(256)
void gemm_wmma(const float* __restrict__ A, const float* __restrict__ B,
               const float* __restrict__ bias, const float* __restrict__ R,
               float* __restrict__ C,
               int M, int N, int K, int lda, int ldb, int ldc, int ldr,
               int act, long sA, long sB, long sC, long sBias)
{
  __shared__ alignas(16) __bf16 As[2][128 * LDSK];
  __shared__ alignas(16) __bf16 Bs[2][128 * LDSK];

  const int tid  = threadIdx.x;
  const int wave = tid >> 5;
  const int lane = tid & 31;
  const int half = lane >> 4;   // 0 or 1
  const int lr   = lane & 15;
  const int wm   = wave >> 2;   // 0..1 -> 64 rows each
  const int wn   = wave & 3;    // 0..3 -> 32 cols each
  const int row0 = blockIdx.y * 128;
  const int col0 = blockIdx.x * 128;
  const int z    = blockIdx.z;

  A += (long)z * sA;
  B += (long)z * sB;
  C += (long)z * sC;
  const float* biasz = bias ? bias + (long)z * sBias : nullptr;

  const bool fullM = (row0 + 128 <= M);   // uniform per block
  const bool fullN = (col0 + 128 <= N);   // uniform per block

  float4 ra[4], rb[4];   // staged tile (tile t+1) while computing tile t

  // ---- load a 128x32 A tile into registers (branch-free edge handling) ----
  auto loadA = [&](int k0) {
#pragma unroll
    for (int i = 0; i < 4; i++) {
      int e = tid + i * 256;
      int r = e >> 3, c = (e & 7) * 4;
      int gr = row0 + r, gc = k0 + c;
      if (fullM) {
        ra[i] = *(const float4*)(A + (long)gr * lda + gc);
      } else {
        bool ok = gr < M;
        long base = ok ? ((long)gr * lda + gc) : 0;
        float x0 = A[base], x1 = A[base + 1], x2 = A[base + 2], x3 = A[base + 3];
        if (!ok) { x0 = x1 = x2 = x3 = 0.f; }
        ra[i] = make_float4(x0, x1, x2, x3);
      }
    }
  };

  // ---- load a 32x128 / 128x32 B tile into registers ----
  auto loadB = [&](int k0) {
#pragma unroll
    for (int i = 0; i < 4; i++) {
      int e = tid + i * 256;
      if (TRANSB) {                       // B[N,K]: read along k
        int r = e >> 3, c = (e & 7) * 4;
        int gn = col0 + r, gc = k0 + c;
        if (fullN) {
          rb[i] = *(const float4*)(B + (long)gn * ldb + gc);
        } else {
          bool ok = gn < N;
          long base = ok ? ((long)gn * ldb + gc) : 0;
          float x0 = B[base], x1 = B[base + 1], x2 = B[base + 2], x3 = B[base + 3];
          if (!ok) { x0 = x1 = x2 = x3 = 0.f; }
          rb[i] = make_float4(x0, x1, x2, x3);
        }
      } else {                            // B[K,N]: read along n (transpose on store)
        int k = e >> 5, n = (e & 31) * 4;
        int gn = col0 + n, gc = k0 + k;
        if (fullN) {
          rb[i] = *(const float4*)(B + (long)gc * ldb + gn);
        } else {
          float x[4];
#pragma unroll
          for (int j = 0; j < 4; j++) {
            bool ok = (gn + j) < N;
            long idx = ok ? ((long)gc * ldb + gn + j) : 0;
            float t = B[idx];
            x[j] = ok ? t : 0.f;
          }
          rb[i] = make_float4(x[0], x[1], x[2], x[3]);
        }
      }
    }
  };

  // ---- convert staged registers to bf16 and store into LDS buffer ----
  auto stage = [&](int dst) {
#pragma unroll
    for (int i = 0; i < 4; i++) {
      int e = tid + i * 256;
      {
        int r = e >> 3, c = (e & 7) * 4;
        Pack4 cv;
        cv.h[0] = (__bf16)ra[i].x; cv.h[1] = (__bf16)ra[i].y;
        cv.h[2] = (__bf16)ra[i].z; cv.h[3] = (__bf16)ra[i].w;
        *(ushort4*)&As[dst][r * LDSK + c] = cv.s;
      }
      if (TRANSB) {
        int r = e >> 3, c = (e & 7) * 4;
        Pack4 cv;
        cv.h[0] = (__bf16)rb[i].x; cv.h[1] = (__bf16)rb[i].y;
        cv.h[2] = (__bf16)rb[i].z; cv.h[3] = (__bf16)rb[i].w;
        *(ushort4*)&Bs[dst][r * LDSK + c] = cv.s;
      } else {
        int k = e >> 5, n = (e & 31) * 4;
        Bs[dst][(n + 0) * LDSK + k] = (__bf16)rb[i].x;
        Bs[dst][(n + 1) * LDSK + k] = (__bf16)rb[i].y;
        Bs[dst][(n + 2) * LDSK + k] = (__bf16)rb[i].z;
        Bs[dst][(n + 3) * LDSK + k] = (__bf16)rb[i].w;
      }
    }
  };

  f32x8 acc[4][2] = {};
  const int nk = K >> 5;          // K % 32 == 0 at every call site

  loadA(0); loadB(0); stage(0);
  int buf = 0;
  for (int t = 0; t < nk; t++) {
    __syncthreads();
    const bool more = (t + 1 < nk);
    if (more) { loadA((t + 1) * 32); loadB((t + 1) * 32); }
    if (t + 2 < nk) {             // gfx1250 global_prefetch_b8, two tiles ahead
      int pr = row0 + (tid & 127); if (pr >= M) pr = M - 1;
      __builtin_prefetch(A + (long)pr * lda + (t + 2) * 32, 0, 1);
    }

    // ---- build fragments from LDS[buf] (ISA 7.12.2 16-bit layouts) ----
    FragU fa[4];
#pragma unroll
    for (int s = 0; s < 4; s++) {
      int m = wm * 64 + s * 16 + lr;
      const __bf16* p = &As[buf][m * LDSK + half * 8];
      fa[s].u[0] = *(const u32x4*)(p);        // K = half*8 .. +7
      fa[s].u[1] = *(const u32x4*)(p + 16);   // K = 16+half*8 .. +7
    }
    FragU fb[2];
#pragma unroll
    for (int t2 = 0; t2 < 2; t2++) {
      int n = wn * 32 + t2 * 16 + lr;
      const __bf16* p = &Bs[buf][n * LDSK + half * 8];
      fb[t2].u[0] = *(const u32x4*)(p);       // K = half*8 .. +7
      fb[t2].u[1] = *(const u32x4*)(p + 16);  // K = 16+half*8 .. +7
    }

#pragma unroll
    for (int s = 0; s < 4; s++)
#pragma unroll
      for (int t2 = 0; t2 < 2; t2++)
        acc[s][t2] = __builtin_amdgcn_wmma_f32_16x16x32_bf16(
            false, fa[s].v, false, fb[t2].v, (short)0, acc[s][t2], false, false);

    if (more) stage(buf ^ 1);     // fill the other buffer (no extra barrier)
    buf ^= 1;
  }

  // ---- epilogue: bias + residual + activation, bounds-checked ----
#pragma unroll
  for (int s = 0; s < 4; s++) {
#pragma unroll
    for (int t2 = 0; t2 < 2; t2++) {
      int col = col0 + wn * 32 + t2 * 16 + lr;
#pragma unroll
      for (int r8 = 0; r8 < 8; r8++) {
        int row = row0 + wm * 64 + s * 16 + half * 8 + r8;
        if (row < M && col < N) {
          float v = acc[s][t2][r8];
          if (biasz) v += biasz[col];
          if (R)     v += R[(long)row * ldr + col];
          if (act == ACT_RELU)     v = fmaxf(v, 0.f);
          else if (act == ACT_SIG) v = 1.f / (1.f + __expf(-v));
          C[(long)row * ldc + col] = v;
        }
      }
    }
  }
}

// ---------------------------------------------------------------------------
// Elementwise / glue kernels
// ---------------------------------------------------------------------------
__global__ void slice_comments_kernel(const float* __restrict__ c, float* __restrict__ o)
{
  long i = blockIdx.x * 256l + threadIdx.x;
  if (i >= 5120l * 768) return;
  int d = (int)(i % 768); long r = i / 768;
  int b = (int)(r / 20), t = (int)(r % 20);
  o[i] = c[((long)b * 50 + t) * 768 + d];
}

__global__ void bcast_asr_kernel(const float* __restrict__ a, float* __restrict__ vfea)
{
  long i = blockIdx.x * 256l + threadIdx.x;
  if (i >= 256l * 64 * 512) return;
  int j = (int)(i & 511); long bt = i >> 9; int b = (int)(bt >> 6);
  vfea[bt * 1024 + 512 + j] = a[(long)b * 512 + j];
}

__global__ void mean20_kernel(const float* __restrict__ x, float* __restrict__ y)
{
  long i = blockIdx.x * 256l + threadIdx.x;
  if (i >= 256l * 1024) return;
  int b = (int)(i >> 10), j = (int)(i & 1023);
  float s = 0.f;
  for (int t = 0; t < 20; t++) s += x[((long)b * 20 + t) * 1024 + j];
  y[i] = s * 0.05f;
}

__global__ void diff_cat_kernel(const float* __restrict__ pv, const float* __restrict__ cv,
                                float* __restrict__ p1in, float* __restrict__ diff,
                                float* __restrict__ fea7)
{
  long i = blockIdx.x * 256l + threadIdx.x;
  if (i >= 256l * 1024) return;
  int b = (int)(i >> 10), j = (int)(i & 1023);
  float a = pv[i], c = cv[i], d = a - c;
  p1in[(long)b * 3072 + j]        = a;
  p1in[(long)b * 3072 + 1024 + j] = c;
  p1in[(long)b * 3072 + 2048 + j] = fabsf(d);
  diff[i] = d;
  fea7[(long)b * 3584 + j]        = a;
  fea7[(long)b * 3584 + 1024 + j] = c;
}

__global__ void wmul_kernel(const float* __restrict__ w, const float* __restrict__ d,
                            float* __restrict__ fea7)
{
  long i = blockIdx.x * 256l + threadIdx.x;
  if (i >= 256l * 1024) return;
  int b = (int)(i >> 10), j = (int)(i & 1023);
  fea7[(long)b * 3584 + 2048 + j] = w[i] * d[i];
}

// ---------------------------------------------------------------------------
// Attention cores (per (batch, head) block)
// ---------------------------------------------------------------------------
__global__ __launch_bounds__(64)
void attn1_kernel(const float* __restrict__ q, const float* __restrict__ k,
                  const float* __restrict__ v, float* __restrict__ out)
{
  int b = blockIdx.x >> 3, h = blockIdx.x & 7;
  int tid = threadIdx.x;
  __shared__ float qs[128], sc[64];
  qs[tid]      = q[(long)b * 1024 + h * 128 + tid];
  qs[tid + 64] = q[(long)b * 1024 + h * 128 + 64 + tid];
  __syncthreads();
  const float* kp = k + ((long)b * 64 + tid) * 1024 + h * 128;
  float s = 0.f;
  for (int d = 0; d < 128; d++) s += qs[d] * kp[d];
  sc[tid] = s * 0.08838834764831845f;   // 1/sqrt(128)
  __syncthreads();
  float m = -1e30f;
  for (int t = 0; t < 64; t++) m = fmaxf(m, sc[t]);
  float e = __expf(sc[tid] - m);
  __syncthreads();
  sc[tid] = e;
  __syncthreads();
  float sum = 0.f;
  for (int t = 0; t < 64; t++) sum += sc[t];
  float inv = 1.f / sum;
  for (int rep = 0; rep < 2; rep++) {
    int d = tid + rep * 64;
    float o = 0.f;
    for (int t = 0; t < 64; t++) o += sc[t] * v[((long)b * 64 + t) * 1024 + h * 128 + d];
    out[(long)b * 1024 + h * 128 + d] = o * inv;
  }
}

__global__ __launch_bounds__(64)
void attn2_kernel(const float* __restrict__ q, const float* __restrict__ k,
                  const float* __restrict__ v, float* __restrict__ out)
{
  int b = blockIdx.x >> 3, h = blockIdx.x & 7;
  int tid = threadIdx.x;
  __shared__ float qs[20 * 128];
  __shared__ float sc[20 * 64];
  for (int i = tid; i < 20 * 128; i += 64) {
    int qi = i >> 7, d = i & 127;
    qs[i] = q[((long)b * 20 + qi) * 1024 + h * 128 + d];
  }
  __syncthreads();
  float accq[20];
#pragma unroll
  for (int i = 0; i < 20; i++) accq[i] = 0.f;
  const float* kp = k + ((long)b * 64 + tid) * 1024 + h * 128;
  for (int d = 0; d < 128; d++) {
    float kv = kp[d];
#pragma unroll
    for (int i = 0; i < 20; i++) accq[i] += qs[i * 128 + d] * kv;
  }
#pragma unroll
  for (int i = 0; i < 20; i++) sc[i * 64 + tid] = accq[i] * 0.08838834764831845f;
  __syncthreads();
  if (tid < 20) {
    float m = -1e30f;
    for (int t = 0; t < 64; t++) m = fmaxf(m, sc[tid * 64 + t]);
    float s = 0.f;
    for (int t = 0; t < 64; t++) { float e = __expf(sc[tid * 64 + t] - m); sc[tid * 64 + t] = e; s += e; }
    float inv = 1.f / s;
    for (int t = 0; t < 64; t++) sc[tid * 64 + t] *= inv;
  }
  __syncthreads();
  for (int i = tid; i < 20 * 128; i += 64) {
    int qi = i >> 7, d = i & 127;
    float o = 0.f;
    for (int t = 0; t < 64; t++) o += sc[qi * 64 + t] * v[((long)b * 64 + t) * 1024 + h * 128 + d];
    out[((long)b * 20 + qi) * 1024 + h * 128 + d] = o;
  }
}

// ---------------------------------------------------------------------------
// MoE: gating (top-4 of 16), combine, cv^2 load-balance loss
// ---------------------------------------------------------------------------
__global__ __launch_bounds__(32)
void gate_topk_kernel(const float* __restrict__ x, const float* __restrict__ wg,
                      float* __restrict__ gates)
{
  int b = blockIdx.x, tid = threadIdx.x;
  __shared__ float lg[16];
  if (tid < 16) {
    float s = 0.f;
    const float* xp = x + (long)b * 1536;
    for (int i = 0; i < 1536; i++) s += xp[i] * wg[i * 16 + tid];
    lg[tid] = s;
  }
  __syncthreads();
  if (tid == 0) {
    float tmp[16];
    for (int e = 0; e < 16; e++) tmp[e] = lg[e];
    float vals[4]; int idx[4];
    for (int kk = 0; kk < 4; kk++) {
      int bi = 0; float bv = -1e30f;
      for (int e = 0; e < 16; e++) if (tmp[e] > bv) { bv = tmp[e]; bi = e; }
      vals[kk] = bv; idx[kk] = bi; tmp[bi] = -1e30f;
    }
    float m = vals[0], s = 0.f, ex[4];
    for (int kk = 0; kk < 4; kk++) { ex[kk] = __expf(vals[kk] - m); s += ex[kk]; }
    for (int e = 0; e < 16; e++) gates[b * 16 + e] = 0.f;
    for (int kk = 0; kk < 4; kk++) gates[b * 16 + idx[kk]] = ex[kk] / s;
  }
}

__global__ void moe_combine_kernel(const float* __restrict__ gates,
                                   const float* __restrict__ y, float* __restrict__ out)
{
  long i = blockIdx.x * 256l + threadIdx.x;
  if (i >= 256l * 512) return;
  int b = (int)(i / 512), n = (int)(i % 512);
  float s = 0.f;
  for (int e = 0; e < 16; e++) {
    float gv = gates[b * 16 + e];
    if (gv != 0.f) s += gv * y[((long)e * 256 + b) * 512 + n];
  }
  out[i] = s;
}

__global__ __launch_bounds__(32)
void moe_loss_kernel(const float* __restrict__ gates, float* __restrict__ loss)
{
  __shared__ float imp[16];
  int tid = threadIdx.x;
  if (tid < 16) {
    float s = 0.f;
    for (int b = 0; b < 256; b++) s += gates[b * 16 + tid];
    imp[tid] = s;
  }
  __syncthreads();
  if (tid == 0) {
    float m = 0.f;
    for (int e = 0; e < 16; e++) m += imp[e];
    m *= (1.f / 16.f);
    float v = 0.f;
    for (int e = 0; e < 16; e++) { float d = imp[e] - m; v += d * d; }
    v *= (1.f / 16.f);
    *loss = v / (m * m + 1e-10f) * 0.01f;
  }
}

// ---------------------------------------------------------------------------
// LayerNorm(+ReLU) over rows of length 256
// ---------------------------------------------------------------------------
__global__ __launch_bounds__(256)
void ln_relu_kernel(const float* __restrict__ x, const float* __restrict__ g,
                    const float* __restrict__ bb, float* __restrict__ y)
{
  const int D = 256;
  int row = blockIdx.x, tid = threadIdx.x;
  __shared__ float red[256];
  float v = x[(long)row * D + tid];
  red[tid] = v; __syncthreads();
  for (int s = 128; s > 0; s >>= 1) { if (tid < s) red[tid] += red[tid + s]; __syncthreads(); }
  float mean = red[0] * (1.f / D);
  __syncthreads();
  float dv = v - mean;
  red[tid] = dv * dv; __syncthreads();
  for (int s = 128; s > 0; s >>= 1) { if (tid < s) red[tid] += red[tid + s]; __syncthreads(); }
  float var = red[0] * (1.f / D);
  float o = dv * rsqrtf(var + 1e-5f) * g[tid] + bb[tid];
  y[(long)row * D + tid] = fmaxf(o, 0.f);
}

// ---------------------------------------------------------------------------
// GRU: sequential scan over the 256-step "sequence" (one layer per launch)
// gi [256,1536] precomputed; whh [1536,512]; out [256,512]
// ---------------------------------------------------------------------------
__global__ __launch_bounds__(512)
void gru_scan_kernel(const float* __restrict__ gi, const float* __restrict__ whh,
                     const float* __restrict__ bhh, float* __restrict__ out)
{
  int j = threadIdx.x;
  __shared__ float hs[512];
  hs[j] = 0.f;
  __syncthreads();
  const float* wr = whh + (long)j * 512;
  const float* wz = whh + (long)(512 + j) * 512;
  const float* wn = whh + (long)(1024 + j) * 512;
  float br = bhh[j], bz = bhh[512 + j], bn = bhh[1024 + j];
  for (int t = 0; t < 256; t++) {
    float sr = br, sz = bz, sn = bn;
    for (int d = 0; d < 512; d++) {
      float hv = hs[d];
      sr += wr[d] * hv; sz += wz[d] * hv; sn += wn[d] * hv;
    }
    const float* g = gi + (long)t * 1536;
    float r = 1.f / (1.f + __expf(-(g[j] + sr)));
    float z = 1.f / (1.f + __expf(-(g[512 + j] + sz)));
    float n = tanhf(g[1024 + j] + r * sn);
    float hnew = (1.f - z) * n + z * hs[j];
    __syncthreads();
    hs[j] = hnew;
    out[(long)t * 512 + j] = hnew;
    __syncthreads();
  }
}

// ---------------------------------------------------------------------------
// Host side
// ---------------------------------------------------------------------------
static inline void gemm(hipStream_t st, const float* A, const float* B, const float* bias,
                        const float* R, float* C, int M, int N, int K,
                        int lda, int ldb, int ldc, int ldr, int act, bool transB,
                        int batch = 1, long sA = 0, long sB = 0, long sC = 0, long sBias = 0)
{
  dim3 grid((N + 127) / 128, (M + 127) / 128, batch), block(256);
  if (transB)
    gemm_wmma<1><<<grid, block, 0, st>>>(A, B, bias, R, C, M, N, K, lda, ldb, ldc, ldr, act, sA, sB, sC, sBias);
  else
    gemm_wmma<0><<<grid, block, 0, st>>>(A, B, bias, R, C, M, N, K, lda, ldb, ldc, ldr, act, sA, sB, sC, sBias);
}

static inline int ewg(long n) { return (int)((n + 255) / 256); }

// workspace layout (floats)
static constexpr long OF_VFEA  = 0;
static constexpr long OF_KH    = OF_VFEA  + 16384l * 1024;
static constexpr long OF_VH    = OF_KH    + 16384l * 1024;
static constexpr long OF_ASR   = OF_VH    + 16384l * 1024;
static constexpr long OF_PFEA2 = OF_ASR   + 256l * 512;
static constexpr long OF_CSL   = OF_PFEA2 + 256l * 1024;
static constexpr long OF_CP    = OF_CSL   + 5120l * 768;
static constexpr long OF_QH1   = OF_CP    + 5120l * 1024;
static constexpr long OF_ATT1  = OF_QH1   + 256l * 1024;
static constexpr long OF_PV    = OF_ATT1  + 256l * 1024;
static constexpr long OF_QH2   = OF_PV    + 256l * 1024;
static constexpr long OF_ATT2  = OF_QH2   + 5120l * 1024;
static constexpr long OF_O2    = OF_ATT2  + 5120l * 1024;
static constexpr long OF_CV    = OF_O2    + 5120l * 1024;
static constexpr long OF_P1IN  = OF_CV    + 256l * 1024;
static constexpr long OF_DIFF  = OF_P1IN  + 256l * 3072;
static constexpr long OF_WSIG  = OF_DIFF  + 256l * 1024;
static constexpr long OF_D2B   = OF_WSIG  + 256l * 1024;
static constexpr long OF_FEA7  = OF_D2B   + 256l * 1024;
static constexpr long OF_FEA   = OF_FEA7  + 256l * 3584;
static constexpr long OF_GATES = OF_FEA   + 256l * 1536;
static constexpr long OF_HALL  = OF_GATES + 256l * 16;
static constexpr long OF_YALL  = OF_HALL  + 16l * 256 * 1024;
static constexpr long OF_CLSF  = OF_YALL  + 16l * 256 * 512;
static constexpr long OF_REGF  = OF_CLSF  + 256l * 512;
static constexpr long OF_CLSH  = OF_REGF  + 256l * 512;
static constexpr long OF_CLSH2 = OF_CLSH  + 256l * 256;
static constexpr long OF_GI    = OF_CLSH2 + 256l * 256;
static constexpr long OF_HA    = OF_GI    + 256l * 1536;
static constexpr long OF_HB    = OF_HA    + 256l * 512;
static constexpr long OF_GFC   = OF_HB    + 256l * 512;
static constexpr long OF_REGH  = OF_GFC   + 256l * 512;
static constexpr long OF_REGH2 = OF_REGH  + 256l * 256;

extern "C" void kernel_launch(void* const* d_in, const int* in_sizes, int n_in,
                              void* d_out, int out_size, void* d_ws, size_t ws_size,
                              hipStream_t stream)
{
  (void)in_sizes; (void)n_in; (void)out_size; (void)ws_size;
  const float* video   = (const float*)d_in[0];
  const float* asr     = (const float*)d_in[1];
  const float* topics  = (const float*)d_in[2];
  const float* title   = (const float*)d_in[3];
  const float* author  = (const float*)d_in[4];
  const float* comment = (const float*)d_in[5];
  auto P = [&](int i) { return (const float*)d_in[6 + i]; };
  // param index map (setup_inputs dict insertion order):
  // 0/1 video_w/b  2/3 asr  4/5 title  6/7 author  8/9 topics1  10/11 comment
  // 12..15 a1 wq,wk,wv,wo  16..19 a1 bq,bk,bv,bo
  // 20..23 a2 wq,wk,wv,wo  24..27 a2 bq,bk,bv,bo
  // 28/29 proj1  30/31 proj2  32/33 reduce1
  // 34 gate_c 35 ec_w1 36 ec_b1 37 ec_w2 38 ec_b2
  // 39 gate_r 40 er_w1 41 er_b1 42 er_w2 43 er_b2
  // 44..49 cls w1,b1,lng,lnb,w2,b2   50..55 reg ...
  // 56.. gru (wih,whh,bih,bhh)*3    68/69 gru_fc w,b
  float* ws  = (float*)d_ws;
  float* out = (float*)d_out;
  hipStream_t st = stream;

  // ---- modality projections ----
  slice_comments_kernel<<<ewg(5120l * 768), 256, 0, st>>>(comment, ws + OF_CSL);
  gemm(st, video, P(0), P(1), nullptr, ws + OF_VFEA, 16384, 512, 768, 768, 768, 1024, 0, ACT_RELU, true);
  gemm(st, asr,   P(2), P(3), nullptr, ws + OF_ASR,    256, 512, 768, 768, 768,  512, 0, ACT_RELU, true);
  bcast_asr_kernel<<<ewg(256l * 64 * 512), 256, 0, st>>>(ws + OF_ASR, ws + OF_VFEA);
  gemm(st, title,  P(4), P(5), nullptr, ws + OF_PFEA2,       256, 512, 768, 768, 768, 1024, 0, ACT_RELU, true);
  gemm(st, author, P(6), P(7), nullptr, ws + OF_PFEA2 + 512, 256, 512, 768, 768, 768, 1024, 0, ACT_RELU, true);
  gemm(st, ws + OF_CSL, P(10), P(11), nullptr, ws + OF_CP, 5120, 1024, 768, 768, 768, 1024, 0, ACT_RELU, true);

  // ---- attention 1 (query = title|author) ----
  gemm(st, ws + OF_PFEA2, P(12), P(16), nullptr, ws + OF_QH1,   256, 1024, 1024, 1024, 1024, 1024, 0, ACT_NONE, true);
  gemm(st, ws + OF_VFEA,  P(13), P(17), nullptr, ws + OF_KH,  16384, 1024, 1024, 1024, 1024, 1024, 0, ACT_NONE, true);
  gemm(st, ws + OF_VFEA,  P(14), P(18), nullptr, ws + OF_VH,  16384, 1024, 1024, 1024, 1024, 1024, 0, ACT_NONE, true);
  attn1_kernel<<<256 * 8, 64, 0, st>>>(ws + OF_QH1, ws + OF_KH, ws + OF_VH, ws + OF_ATT1);
  gemm(st, ws + OF_ATT1, P(15), P(19), ws + OF_PFEA2, ws + OF_PV, 256, 1024, 1024, 1024, 1024, 1024, 1024, ACT_NONE, true);

  // ---- attention 2 (queries = 20 comments), reuse KH/VH ----
  gemm(st, ws + OF_CP,   P(20), P(24), nullptr, ws + OF_QH2, 5120, 1024, 1024, 1024, 1024, 1024, 0, ACT_NONE, true);
  gemm(st, ws + OF_VFEA, P(21), P(25), nullptr, ws + OF_KH, 16384, 1024, 1024, 1024, 1024, 1024, 0, ACT_NONE, true);
  gemm(st, ws + OF_VFEA, P(22), P(26), nullptr, ws + OF_VH, 16384, 1024, 1024, 1024, 1024, 1024, 0, ACT_NONE, true);
  attn2_kernel<<<256 * 8, 64, 0, st>>>(ws + OF_QH2, ws + OF_KH, ws + OF_VH, ws + OF_ATT2);
  gemm(st, ws + OF_ATT2, P(23), P(27), ws + OF_CP, ws + OF_O2, 5120, 1024, 1024, 1024, 1024, 1024, 1024, ACT_NONE, true);
  mean20_kernel<<<ewg(256l * 1024), 256, 0, st>>>(ws + OF_O2, ws + OF_CV);

  // ---- gated fusion ----
  diff_cat_kernel<<<ewg(256l * 1024), 256, 0, st>>>(ws + OF_PV, ws + OF_CV, ws + OF_P1IN, ws + OF_DIFF, ws + OF_FEA7);
  gemm(st, ws + OF_P1IN, P(28), P(29), nullptr, ws + OF_WSIG, 256, 1024, 3072, 3072, 3072, 1024, 0, ACT_SIG,  true);
  gemm(st, ws + OF_DIFF, P(30), P(31), nullptr, ws + OF_D2B,  256, 1024, 1024, 1024, 1024, 1024, 0, ACT_NONE, true);
  wmul_kernel<<<ewg(256l * 1024), 256, 0, st>>>(ws + OF_WSIG, ws + OF_D2B, ws + OF_FEA7);
  gemm(st, topics, P(8), P(9), nullptr, ws + OF_FEA7 + 3072, 256, 512, 768, 768, 768, 3584, 0, ACT_RELU, true);
  gemm(st, ws + OF_FEA7, P(32), P(33), nullptr, ws + OF_FEA, 256, 1536, 3584, 3584, 3584, 1536, 0, ACT_NONE, true);

  // ---- MoE branches (c then r, reusing expert buffers) ----
  const int gIdx[2]  = {34, 39};
  const long dstO[2] = {OF_CLSF, OF_REGF};
  for (int br = 0; br < 2; br++) {
    int g0 = gIdx[br];
    gate_topk_kernel<<<256, 32, 0, st>>>(ws + OF_FEA, P(g0), ws + OF_GATES);
    // h = relu(x @ w1[e] + b1[e])   [16][256][1024]
    gemm(st, ws + OF_FEA, P(g0 + 1), P(g0 + 2), nullptr, ws + OF_HALL,
         256, 1024, 1536, 1536, 1024, 1024, 0, ACT_RELU, false,
         16, 0, 1536l * 1024, 256l * 1024, 1024);
    // y = h @ w2[e] + b2[e]         [16][256][512]
    gemm(st, ws + OF_HALL, P(g0 + 3), P(g0 + 4), nullptr, ws + OF_YALL,
         256, 512, 1024, 1024, 512, 512, 0, ACT_NONE, false,
         16, 256l * 1024, 1024l * 512, 256l * 512, 512);
    moe_combine_kernel<<<ewg(256l * 512), 256, 0, st>>>(ws + OF_GATES, ws + OF_YALL, ws + dstO[br]);
    moe_loss_kernel<<<1, 32, 0, st>>>(ws + OF_GATES, out + 768 + br);
  }

  // ---- classifier head -> out[0:512) ----
  gemm(st, ws + OF_CLSF, P(44), P(45), nullptr, ws + OF_CLSH, 256, 256, 512, 512, 512, 256, 0, ACT_NONE, true);
  ln_relu_kernel<<<256, 256, 0, st>>>(ws + OF_CLSH, P(46), P(47), ws + OF_CLSH2);
  gemm(st, ws + OF_CLSH2, P(48), P(49), nullptr, out, 256, 2, 256, 256, 256, 2, 0, ACT_NONE, true);

  // ---- GRU (3 layers over the 256-step "sequence") ----
  gemm(st, ws + OF_REGF, P(56), P(58), nullptr, ws + OF_GI, 256, 1536, 512, 512, 512, 1536, 0, ACT_NONE, true);
  gru_scan_kernel<<<1, 512, 0, st>>>(ws + OF_GI, P(57), P(59), ws + OF_HA);
  gemm(st, ws + OF_HA, P(60), P(62), nullptr, ws + OF_GI, 256, 1536, 512, 512, 512, 1536, 0, ACT_NONE, true);
  gru_scan_kernel<<<1, 512, 0, st>>>(ws + OF_GI, P(61), P(63), ws + OF_HB);
  gemm(st, ws + OF_HB, P(64), P(66), nullptr, ws + OF_GI, 256, 1536, 512, 512, 512, 1536, 0, ACT_NONE, true);
  gru_scan_kernel<<<1, 512, 0, st>>>(ws + OF_GI, P(65), P(67), ws + OF_HA);
  gemm(st, ws + OF_HA, P(68), P(69), nullptr, ws + OF_GFC, 256, 512, 512, 512, 512, 512, 0, ACT_NONE, true);

  // ---- regressor head -> out[512:768) ----
  gemm(st, ws + OF_GFC, P(50), P(51), nullptr, ws + OF_REGH, 256, 256, 512, 512, 512, 256, 0, ACT_NONE, true);
  ln_relu_kernel<<<256, 256, 0, st>>>(ws + OF_REGH, P(52), P(53), ws + OF_REGH2);
  gemm(st, ws + OF_REGH2, P(54), P(55), nullptr, out + 512, 256, 1, 256, 256, 256, 1, 0, ACT_NONE, true);
}
